// FormulaGINE_Sem_4595615007316
// MI455X (gfx1250) — matrix operand order repaired and checked by reading, not compiled, not run on hardware
//
#include <hip/hip_runtime.h>
#include <hip/hip_bf16.h>

#define N_NODES   100000
#define N_EDGES   300000
#define NUM_GRAPHS 128
#define HID       256
#define SEM_IN    1536
#define MAX_CHILD 200

typedef __attribute__((ext_vector_type(16))) __bf16 v16bf;
typedef __attribute__((ext_vector_type(8)))  float  v8f;

// ---------- helpers ----------

__device__ __forceinline__ unsigned short f2bf_bits(float f) {
  union { float f; unsigned u; } x; x.f = f;
  unsigned r = x.u + 0x7FFFu + ((x.u >> 16) & 1u);   // round-to-nearest-even
  return (unsigned short)(r >> 16);
}

union V16U { v16bf v; uint4 q[2]; };

__device__ __forceinline__ v16bf ld_frag(const unsigned short* p0, const unsigned short* p1) {
  V16U u;
  u.q[0] = *(const uint4*)p0;   // 16B chunk (8 bf16)
  u.q[1] = *(const uint4*)p1;   // 16B chunk (8 bf16)
  return u.v;
}

__device__ __forceinline__ void atomAddF(float* p, float v) {
  __hip_atomic_fetch_add(p, v, __ATOMIC_RELAXED, __HIP_MEMORY_SCOPE_AGENT);
}

// ---------- weight convert: W[K,N] f32 -> Wt[N,K] bf16 ----------

__global__ __launch_bounds__(256) void wconv_kernel(const float* __restrict__ W,
                                                    unsigned short* __restrict__ Wt,
                                                    int K, int N) {
  int idx = blockIdx.x * 256 + threadIdx.x;
  if (idx >= K * N) return;
  int k = idx / N;
  int n = idx - k * N;
  Wt[n * K + k] = f2bf_bits(W[idx]);
}

// ---------- zero fill ----------

__global__ __launch_bounds__(256) void zero_kernel(float* __restrict__ p, int n) {
  int i = blockIdx.x * 256 + threadIdx.x;
  if (i < n) p[i] = 0.f;
}

// ---------- fused WMMA GEMM: out = act((A [+A2]) @ Wt^T + bias [+ label/type emb]) ----------
// One block: 16 rows x 256 cols. 8 waves, each wave owns two 16x16 col tiles.
// A is fp32 row-major, staged per K-step into a double-buffered LDS bf16 tile
// (row stride 40 shorts = conflict-free, 16B aligned). One barrier per K-step;
// staging of step k+1 overlaps the WMMAs of step k.
// Wt is bf16, N-major [256, KDIM] so B fragments are 32B contiguous per lane.

template<int KDIM, bool HAS_A2, bool HAS_EMB, bool RELU>
__global__ __launch_bounds__(256) void gemm_wmma_256(
    const float* __restrict__ A, const float* __restrict__ A2,
    const unsigned short* __restrict__ Wt, const float* __restrict__ bias,
    const int* __restrict__ x, const float* __restrict__ label_emb,
    const float* __restrict__ type_emb,
    float* __restrict__ out)
{
  __shared__ unsigned short sA[2][16 * 40];
  const int tid  = threadIdx.x;
  const int wave = tid >> 5;
  const int lane = tid & 31;
  const int row0 = blockIdx.x << 4;
  const int n0a  = wave << 5;
  const int n0b  = n0a + 16;

  // staging assignment: thread stages rows r and r+8, column c, of each 16x32 tile
  const int r = tid >> 5;           // 0..7
  const int c = tid & 31;
  const float* pa0 = A + (row0 + r) * KDIM + c;
  const float* pa1 = A + (row0 + r + 8) * KDIM + c;
  const float* pq0 = HAS_A2 ? (A2 + (row0 + r) * KDIM + c) : nullptr;
  const float* pq1 = HAS_A2 ? (A2 + (row0 + r + 8) * KDIM + c) : nullptr;

  v8f acc0 = {0.f,0.f,0.f,0.f,0.f,0.f,0.f,0.f};
  v8f acc1 = {0.f,0.f,0.f,0.f,0.f,0.f,0.f,0.f};

  const int arow  = lane & 15;
  const int koffA = (lane < 16) ? 0 : 16;   // A frag: K 0..15 | 16..31 contiguous
  const int koffB = (lane < 16) ? 0 : 16;   // B frag: K 0..15 | 16..31 contiguous
  const int ncol  = lane & 15;

  auto stage = [&](int k0, int buf) {
    float v0 = pa0[k0];
    float v1 = pa1[k0];
    if (HAS_A2) { v0 += pq0[k0]; v1 += pq1[k0]; }
    sA[buf][r * 40 + c]       = f2bf_bits(v0);
    sA[buf][(r + 8) * 40 + c] = f2bf_bits(v1);
  };

  stage(0, 0);
  int cur = 0;

  for (int k0 = 0; k0 < KDIM; k0 += 32) {
    __syncthreads();                         // buf[cur] ready for all waves
    if (k0 + 32 < KDIM) {
      stage(k0 + 32, cur ^ 1);               // overlap with WMMAs below
      if (k0 + 64 < KDIM)
        __builtin_prefetch(pa0 + k0 + 64, 0, 3);   // global_prefetch_b8, next-next tile
    }

    v16bf a = ld_frag(&sA[cur][arow * 40 + koffA], &sA[cur][arow * 40 + koffA + 8]);

    const unsigned short* bp0 = &Wt[(n0a + ncol) * KDIM + k0 + koffB];
    const unsigned short* bp1 = &Wt[(n0b + ncol) * KDIM + k0 + koffB];
    v16bf b0 = ld_frag(bp0, bp0 + 8);
    v16bf b1 = ld_frag(bp1, bp1 + 8);

    acc0 = __builtin_amdgcn_wmma_f32_16x16x32_bf16(false, a, false, b0, (short)0, acc0, false, false);
    acc1 = __builtin_amdgcn_wmma_f32_16x16x32_bf16(false, a, false, b1, (short)0, acc1, false, false);
    cur ^= 1;
  }

  const int moff = (lane >> 4) << 3;   // lanes 16-31 hold M = v+8
#pragma unroll
  for (int v = 0; v < 8; ++v) {
    int m = row0 + v + moff;
    float e0 = acc0[v] + bias[n0a + ncol];
    float e1 = acc1[v] + bias[n0b + ncol];
    if (HAS_EMB) {
      int lb = x[2 * m + 0];
      int tp = x[2 * m + 1];
      e0 += label_emb[lb * HID + n0a + ncol] + type_emb[tp * HID + n0a + ncol];
      e1 += label_emb[lb * HID + n0b + ncol] + type_emb[tp * HID + n0b + ncol];
    }
    if (RELU) { e0 = fmaxf(e0, 0.f); e1 = fmaxf(e1, 0.f); }
    out[m * HID + n0a + ncol] = e0;
    out[m * HID + n0b + ncol] = e1;
  }
}

// ---------- edge message + scatter: agg[dst] += relu(h[src] + role + child) ----------

__global__ __launch_bounds__(256) void edge_msg_kernel(
    const float* __restrict__ h, const int* __restrict__ edge_attr,
    const int* __restrict__ edge_index,
    const float* __restrict__ role_emb, const float* __restrict__ child_emb,
    float* __restrict__ agg)
{
  int idx = blockIdx.x * 256 + threadIdx.x;       // E * 64 quad-slots
  if (idx >= N_EDGES * 64) return;
  int e = idx >> 6;
  int q = (idx & 63) << 2;
  int src = edge_index[e];
  int dst = edge_index[N_EDGES + e];
  int r  = edge_attr[2 * e + 0];
  int ci = edge_attr[2 * e + 1];
  ci = ci < 0 ? 0 : (ci > MAX_CHILD - 1 ? MAX_CHILD - 1 : ci);

  const float4 hv = *(const float4*)&h[src * HID + q];
  const float4 rv = *(const float4*)&role_emb[r * HID + q];
  const float4 cv = *(const float4*)&child_emb[ci * HID + q];
  float m0 = fmaxf(hv.x + rv.x + cv.x, 0.f);
  float m1 = fmaxf(hv.y + rv.y + cv.y, 0.f);
  float m2 = fmaxf(hv.z + rv.z + cv.z, 0.f);
  float m3 = fmaxf(hv.w + rv.w + cv.w, 0.f);
  float* d = &agg[dst * HID + q];
  atomAddF(d + 0, m0);
  atomAddF(d + 1, m1);
  atomAddF(d + 2, m2);
  atomAddF(d + 3, m3);
}

// ---------- global mean pool (batch is sorted -> segmented register sums) ----------

#define POOL_CHUNK 125

__global__ __launch_bounds__(256) void pool_sum_kernel(const float* __restrict__ h,
                                                       const int* __restrict__ batch,
                                                       float* __restrict__ pooled) {
  int col = threadIdx.x;
  int n0 = blockIdx.x * POOL_CHUNK;
  int nend = n0 + POOL_CHUNK; if (nend > N_NODES) nend = N_NODES;
  if (n0 >= N_NODES) return;
  int cur = batch[n0];
  float acc = 0.f;
  for (int n = n0; n < nend; ++n) {
    int g = batch[n];
    if (g != cur) { atomAddF(&pooled[cur * HID + col], acc); acc = 0.f; cur = g; }
    acc += h[n * HID + col];
  }
  atomAddF(&pooled[cur * HID + col], acc);
}

__global__ __launch_bounds__(256) void count_kernel(const int* __restrict__ batch,
                                                    float* __restrict__ counts) {
  int n = blockIdx.x * 256 + threadIdx.x;
  if (n < N_NODES) atomAddF(&counts[batch[n]], 1.f);
}

// ---------- final projection: out[g] = (pooled[g]/cnt) @ proj_W + proj_b ----------

__global__ __launch_bounds__(256) void proj_kernel(const float* __restrict__ pooled,
                                                   const float* __restrict__ counts,
                                                   const float* __restrict__ proj_W,
                                                   const float* __restrict__ proj_b,
                                                   float* __restrict__ out) {
  __shared__ float pr[HID];
  int g = blockIdx.x, o = threadIdx.x;
  float cnt = counts[g];
  cnt = cnt > 1.f ? cnt : 1.f;
  pr[o] = pooled[g * HID + o] / cnt;
  __syncthreads();
  float s = proj_b[o];
  for (int c = 0; c < HID; ++c) s = fmaf(pr[c], proj_W[c * HID + o], s);
  out[g * HID + o] = s;
}

// ---------- launch ----------

extern "C" void kernel_launch(void* const* d_in, const int* in_sizes, int n_in,
                              void* d_out, int out_size, void* d_ws, size_t ws_size,
                              hipStream_t stream) {
  const int*   x         = (const int*)  d_in[0];
  const float* x_sem     = (const float*)d_in[1];
  const int*   edge_attr = (const int*)  d_in[2];
  const int*   edge_idx  = (const int*)  d_in[3];
  const int*   batch     = (const int*)  d_in[4];
  const float* label_emb = (const float*)d_in[5];
  const float* type_emb  = (const float*)d_in[6];
  const float* sem_W     = (const float*)d_in[7];
  const float* sem_b     = (const float*)d_in[8];
  const float* role_emb  = (const float*)d_in[9];
  const float* child_emb = (const float*)d_in[10];
  const float* W1_0 = (const float*)d_in[11]; const float* b1_0 = (const float*)d_in[12];
  const float* W2_0 = (const float*)d_in[13]; const float* b2_0 = (const float*)d_in[14];
  const float* W1_1 = (const float*)d_in[15]; const float* b1_1 = (const float*)d_in[16];
  const float* W2_1 = (const float*)d_in[17]; const float* b2_1 = (const float*)d_in[18];
  const float* proj_W = (const float*)d_in[19]; const float* proj_b = (const float*)d_in[20];
  float* out = (float*)d_out;

  // workspace layout (~308.7 MB total)
  char* ws = (char*)d_ws;
  const size_t NF = (size_t)N_NODES * HID * sizeof(float);   // 102,400,000
  float* h   = (float*)(ws);
  float* agg = (float*)(ws + NF);
  float* z   = (float*)(ws + 2 * NF);
  size_t off = 3 * NF;
  unsigned short* semWt = (unsigned short*)(ws + off); off += (size_t)HID * SEM_IN * 2; // 786,432
  unsigned short* W10t  = (unsigned short*)(ws + off); off += (size_t)HID * HID * 2;
  unsigned short* W20t  = (unsigned short*)(ws + off); off += (size_t)HID * HID * 2;
  unsigned short* W11t  = (unsigned short*)(ws + off); off += (size_t)HID * HID * 2;
  unsigned short* W21t  = (unsigned short*)(ws + off); off += (size_t)HID * HID * 2;
  float* pooled = (float*)(ws + off);                    // NUM_GRAPHS*HID floats
  float* counts = pooled + NUM_GRAPHS * HID;             // NUM_GRAPHS floats (contiguous)

  const int ROW_TILES = N_NODES / 16;   // 6250 (exact)
  const int EDGE_BLKS = (N_EDGES * 64) / 256;
  const int ZERO_BLKS = (N_NODES * HID + 255) / 256;

  // 1) bf16-transpose weights (L2-resident B operands)
  wconv_kernel<<<(SEM_IN * HID + 255) / 256, 256, 0, stream>>>(sem_W, semWt, SEM_IN, HID);
  wconv_kernel<<<(HID * HID + 255) / 256, 256, 0, stream>>>(W1_0, W10t, HID, HID);
  wconv_kernel<<<(HID * HID + 255) / 256, 256, 0, stream>>>(W2_0, W20t, HID, HID);
  wconv_kernel<<<(HID * HID + 255) / 256, 256, 0, stream>>>(W1_1, W11t, HID, HID);
  wconv_kernel<<<(HID * HID + 255) / 256, 256, 0, stream>>>(W2_1, W21t, HID, HID);

  // 2) node embed: h = x_sem @ sem_W + sem_b + label_emb[x0] + type_emb[x1]
  gemm_wmma_256<SEM_IN, false, true, false><<<ROW_TILES, 256, 0, stream>>>(
      x_sem, nullptr, semWt, sem_b, x, label_emb, type_emb, h);

  // 3) GINE layer 0
  zero_kernel<<<ZERO_BLKS, 256, 0, stream>>>(agg, N_NODES * HID);
  edge_msg_kernel<<<EDGE_BLKS, 256, 0, stream>>>(h, edge_attr, edge_idx, role_emb, child_emb, agg);
  gemm_wmma_256<HID, true, false, true><<<ROW_TILES, 256, 0, stream>>>(
      h, agg, W10t, b1_0, nullptr, nullptr, nullptr, z);
  gemm_wmma_256<HID, false, false, false><<<ROW_TILES, 256, 0, stream>>>(
      z, nullptr, W20t, b2_0, nullptr, nullptr, nullptr, h);

  // 4) GINE layer 1
  zero_kernel<<<ZERO_BLKS, 256, 0, stream>>>(agg, N_NODES * HID);
  edge_msg_kernel<<<EDGE_BLKS, 256, 0, stream>>>(h, edge_attr, edge_idx, role_emb, child_emb, agg);
  gemm_wmma_256<HID, true, false, true><<<ROW_TILES, 256, 0, stream>>>(
      h, agg, W11t, b1_1, nullptr, nullptr, nullptr, z);
  gemm_wmma_256<HID, false, false, false><<<ROW_TILES, 256, 0, stream>>>(
      z, nullptr, W21t, b2_1, nullptr, nullptr, nullptr, h);

  // 5) mean pool + projection
  zero_kernel<<<(NUM_GRAPHS * HID + NUM_GRAPHS + 255) / 256, 256, 0, stream>>>(
      pooled, NUM_GRAPHS * HID + NUM_GRAPHS);
  pool_sum_kernel<<<(N_NODES + POOL_CHUNK - 1) / POOL_CHUNK, 256, 0, stream>>>(h, batch, pooled);
  count_kernel<<<(N_NODES + 255) / 256, 256, 0, stream>>>(batch, counts);
  proj_kernel<<<NUM_GRAPHS, 256, 0, stream>>>(pooled, counts, proj_W, proj_b, out);
}
